// MLA_18915035972234
// MI455X (gfx1250) — compile-verified, matrix operand order checked by reading, not run on hardware
//
#include <hip/hip_runtime.h>
#include <hip/hip_bf16.h>
#include <cstddef>
#include <cstdint>

// ---------------------------------------------------------------------------
// MLA prefill for MI455X (gfx1250).
//   * all matmuls on v_wmma_f32_16x16x32_bf16 (f32 accumulate)
//   * all global->LDS tile staging through the Tensor Data Mover
//     (TENSOR_LOAD_TO_LDS + s_wait_tensorcnt), double-buffered
//   * V kept transposed (vT[H][DV][S]) so every WMMA fragment load is a
//     contiguous ds_load_b128 pair
// Shapes hard-coded: B=1 S=2048 HID=4096 H=32 Q_LORA=1536 KV_LORA=512
// NOPE=128 ROPE=64 VDIM=128.
// ---------------------------------------------------------------------------

typedef __attribute__((ext_vector_type(16))) __bf16 v16bf;
typedef __attribute__((ext_vector_type(8)))  float  v8f;
typedef unsigned int u32x4 __attribute__((ext_vector_type(4)));
typedef int          i32x4 __attribute__((ext_vector_type(4)));
typedef int          i32x8 __attribute__((ext_vector_type(8)));

#define S_LEN   2048
#define HID     4096
#define NHEAD   32
#define DQK     192
#define DV      128

// ---------------------------------------------------------------------------
// TDM: 2-D tile load, global (row-major, strided) -> LDS (packed), bf16 data.
// D# layout per CDNA5 ISA (groups 2/3 zero => 2-D).
// Issue from one wave only; completion tracked on that wave's TENSORcnt.
// ---------------------------------------------------------------------------
__device__ __forceinline__ void tdm_load_2d_bf16(void* lds_dst, const __bf16* src,
                                                 unsigned tile_w, unsigned tile_h,
                                                 unsigned row_stride_elems) {
  const unsigned long long ga = (unsigned long long)(uintptr_t)src;
  u32x4 g0;
  g0[0] = 1u;                                                // count=1, user D#
  g0[1] = (unsigned)(uintptr_t)lds_dst;                      // LDS byte address
  g0[2] = (unsigned)ga;                                      // global addr lo
  g0[3] = (unsigned)((ga >> 32) & 0x01FFFFFFu) | (2u << 30); // addr hi | type=2
  i32x8 g1;
  g1[0] = (int)(1u << 16);            // wg_mask=0, data_size=1 (2 bytes)
  g1[1] = (int)(tile_w << 16);        // tensor_dim0[15:0]   (== tile_w, no OOB)
  g1[2] = (int)(tile_h << 16);        // tensor_dim0 hi=0 | tensor_dim1[15:0]
  g1[3] = (int)(tile_w << 16);        // tensor_dim1 hi=0 | tile_dim0
  g1[4] = (int)tile_h;                // tile_dim1 | tile_dim2=0
  g1[5] = (int)row_stride_elems;      // tensor_dim0_stride lo32 (elements)
  g1[6] = 0;
  g1[7] = 0;
  const i32x4 z4 = {0, 0, 0, 0};
#if __clang_major__ >= 23
  const i32x8 z8 = {0, 0, 0, 0, 0, 0, 0, 0};
  __builtin_amdgcn_tensor_load_to_lds(g0, g1, z4, z4, z8, 0);
#else
  __builtin_amdgcn_tensor_load_to_lds(g0, g1, z4, z4, 0);
#endif
}

// ---------------------------------------------------------------------------
// WMMA fragment loader (CDNA5 ISA §7.12.2, 16-bit A 16x32):
//   lane m = lane&15, half = lane>>4
//   dword d: k0 = (d>=4?16:0) + half*8 + 2*(d&3); elements k0, k0+1
// B fragment for C=A*B with B^T row-major == A-style fragment of B^T.
// ---------------------------------------------------------------------------
__device__ __forceinline__ v16bf load_frag(const __bf16* base, int ld, int lane) {
  const int m    = lane & 15;
  const int half = (lane >> 4) & 1;
  const __bf16* row = base + m * ld;
  v16bf f;
#pragma unroll
  for (int d = 0; d < 8; ++d) {
    const int k0 = ((d & 4) << 2) + half * 8 + ((d & 3) << 1);
    f[2 * d]     = row[k0];
    f[2 * d + 1] = row[k0 + 1];
  }
  return f;
}

// ---------------------------------------------------------------------------
// f32 -> bf16 bulk convert (vector loads, packed converts). n % 1024 == 0.
// ---------------------------------------------------------------------------
__global__ __launch_bounds__(256)
void cvt_bf16_kernel(const float* __restrict__ in, __bf16* __restrict__ out, long n) {
  const long i = ((long)blockIdx.x * 256 + threadIdx.x) * 4;
  if (i < n) {
    const float4 v = *(const float4*)(in + i);
    out[i + 0] = (__bf16)v.x;
    out[i + 1] = (__bf16)v.y;
    out[i + 2] = (__bf16)v.z;
    out[i + 3] = (__bf16)v.w;
  }
}

// ---------------------------------------------------------------------------
// GEMM: C[M,N] (f32) = A[M,K] (bf16) * W[N,K]^T (bf16)
// 256x64 block tile, K-step 32, 8 waves; wave w owns a 32x64 strip
// (2 A-fragments x 4 B-fragments = 8 WMMA / K-step, B-frag reused twice).
// TDM-staged + double-buffered LDS.  All call-site dims divide the tiles.
// ---------------------------------------------------------------------------
#define GT_M 256
#define GT_N 64
#define GT_K 32

__global__ __launch_bounds__(256)
void gemm_bf16(const __bf16* __restrict__ A, const __bf16* __restrict__ W,
               float* __restrict__ C, int M, int N, int K) {
  __shared__ __bf16 sA[2][GT_M][GT_K];   // 32 KB
  __shared__ __bf16 sW[2][GT_N][GT_K];   //  8 KB

  const int tid  = threadIdx.x;
  const int wave = tid >> 5;
  const int lane = tid & 31;
  const int m0   = blockIdx.y * GT_M;
  const int n0   = blockIdx.x * GT_N;

  const __bf16* Ablk = A + (size_t)m0 * K;
  const __bf16* Wblk = W + (size_t)n0 * K;
  const int nsteps = K / GT_K;

  if (wave == 0) {
    tdm_load_2d_bf16(&sA[0][0][0], Ablk, GT_K, GT_M, (unsigned)K);
    tdm_load_2d_bf16(&sW[0][0][0], Wblk, GT_K, GT_N, (unsigned)K);
    __builtin_amdgcn_s_wait_tensorcnt(0);
  }
  __syncthreads();

  v8f acc[2][4] = {};

  for (int t = 0; t < nsteps; ++t) {
    const int cur = t & 1;
    const int nxt = cur ^ 1;
    if (wave == 0 && t + 1 < nsteps) {
      tdm_load_2d_bf16(&sA[nxt][0][0], Ablk + (size_t)(t + 1) * GT_K, GT_K, GT_M, (unsigned)K);
      tdm_load_2d_bf16(&sW[nxt][0][0], Wblk + (size_t)(t + 1) * GT_K, GT_K, GT_N, (unsigned)K);
    }

    const v16bf fa0 = load_frag(&sA[cur][wave * 32][0],      GT_K, lane);
    const v16bf fa1 = load_frag(&sA[cur][wave * 32 + 16][0], GT_K, lane);
#pragma unroll
    for (int j = 0; j < 4; ++j) {
      const v16bf fb = load_frag(&sW[cur][j * 16][0], GT_K, lane);
      acc[0][j] = __builtin_amdgcn_wmma_f32_16x16x32_bf16(
          false, fa0, false, fb, (short)0, acc[0][j], false, false);
      acc[1][j] = __builtin_amdgcn_wmma_f32_16x16x32_bf16(
          false, fa1, false, fb, (short)0, acc[1][j], false, false);
    }

    if (wave == 0) __builtin_amdgcn_s_wait_tensorcnt(0);
    __syncthreads();
  }

  const int mrow = (lane >> 4) * 8;
  const int ncol = lane & 15;
#pragma unroll
  for (int a = 0; a < 2; ++a)
#pragma unroll
    for (int j = 0; j < 4; ++j)
#pragma unroll
      for (int r = 0; r < 8; ++r)
        C[(size_t)(m0 + wave * 32 + a * 16 + mrow + r) * N + n0 + j * 16 + ncol] =
            acc[a][j][r];
}

// ---------------------------------------------------------------------------
// RMS norm over a column slice; bf16 output (feeds the next WMMA GEMM).
// ---------------------------------------------------------------------------
__global__ __launch_bounds__(256)
void rmsnorm_kernel(const float* __restrict__ in, int stride, int col0, int width,
                    const float* __restrict__ w, __bf16* __restrict__ out) {
  const int s = blockIdx.x;
  const float* row = in + (size_t)s * stride + col0;
  __shared__ float red[256];
  float ss = 0.f;
  for (int c = threadIdx.x; c < width; c += 256) {
    const float v = row[c];
    ss += v * v;
  }
  red[threadIdx.x] = ss;
  __syncthreads();
  for (int o = 128; o > 0; o >>= 1) {
    if (threadIdx.x < o) red[threadIdx.x] += red[threadIdx.x + o];
    __syncthreads();
  }
  const float inv = rsqrtf(red[0] / (float)width + 1e-6f);
  __bf16* orow = out + (size_t)s * width;
  for (int c = threadIdx.x; c < width; c += 256)
    orow[c] = (__bf16)(row[c] * inv * w[c]);
}

// ---------------------------------------------------------------------------
// RoPE q_pe + pack q into bf16 qf[S][H][192]
// ---------------------------------------------------------------------------
__global__ __launch_bounds__(256)
void prep_qf_kernel(const float* __restrict__ q, const int* __restrict__ pos_ids,
                    __bf16* __restrict__ qf) {
  const int s = blockIdx.x;
  const float pos = (float)pos_ids[s];
  const float* qrow = q + (size_t)s * (NHEAD * DQK);
  __bf16* orow = qf + (size_t)s * (NHEAD * DQK);
  for (int e = threadIdx.x; e < NHEAD * DQK; e += 256) {
    const int h = e / DQK, d = e % DQK;
    const float* qh = qrow + h * DQK;
    float val;
    if (d < 128) {
      val = qh[d];
    } else {
      const int j = d - 128;
      const int i = (j < 32) ? j : (j - 32);
      const float freq = __powf(10000.0f, -(float)(2 * i) / 64.0f);
      const float ang = pos * freq;
      const float c = __cosf(ang), sn = __sinf(ang);
      const float x1 = qh[128 + i], x2 = qh[160 + i];
      val = (j < 32) ? (x1 * c - x2 * sn) : (x2 * c + x1 * sn);
    }
    orow[e] = (__bf16)val;
  }
}

// ---------------------------------------------------------------------------
// RoPE k_pe (shared across heads) + pack kf[S][H][192]
// ---------------------------------------------------------------------------
__global__ __launch_bounds__(256)
void prep_kf_kernel(const float* __restrict__ kv, const float* __restrict__ a,
                    const int* __restrict__ pos_ids, __bf16* __restrict__ kf) {
  const int s = blockIdx.x;
  const float pos = (float)pos_ids[s];
  __shared__ float kpe[64];
  const float* arow = a + (size_t)s * 2112 + 2048;
  if (threadIdx.x < 32) {
    const int i = threadIdx.x;
    const float freq = __powf(10000.0f, -(float)(2 * i) / 64.0f);
    const float ang = pos * freq;
    const float c = __cosf(ang), sn = __sinf(ang);
    const float x1 = arow[i], x2 = arow[32 + i];
    kpe[i]      = x1 * c - x2 * sn;
    kpe[32 + i] = x2 * c + x1 * sn;
  }
  __syncthreads();
  const float* kvrow = kv + (size_t)s * (NHEAD * 256);
  __bf16* krow = kf + (size_t)s * (NHEAD * DQK);
  for (int e = threadIdx.x; e < NHEAD * DQK; e += 256) {
    const int h = e / DQK, d = e % DQK;
    const float val = (d < 128) ? kvrow[h * 256 + d] : kpe[d - 128];
    krow[e] = (__bf16)val;
  }
}

// ---------------------------------------------------------------------------
// V transpose: vT[H][DV][S] (bf16) from kv_buf[S][H][256] f32 (v = cols 128..)
// One block per (h,d) row -> contiguous writes along S.
// ---------------------------------------------------------------------------
__global__ __launch_bounds__(256)
void prep_vT_kernel(const float* __restrict__ kv, __bf16* __restrict__ vT) {
  const int hd = blockIdx.x;            // 0 .. H*DV-1
  const int h = hd / DV, d = hd % DV;
  __bf16* orow = vT + (size_t)hd * S_LEN;
  const float* src = kv + (size_t)h * 256 + 128 + d;
  for (int s = threadIdx.x; s < S_LEN; s += 256)
    orow[s] = (__bf16)src[(size_t)s * (NHEAD * 256)];
}

// ---------------------------------------------------------------------------
// Causal flash attention: one head x 128 query rows per block, 8 waves.
// K (32x192) and V^T (128x32) tiles staged by TDM, double-buffered.
// Scores: 12 WMMA over d=192 x 32 keys; PV: 8 WMMA from row-major V^T tiles.
// P transposed C-layout -> A-layout via per-wave LDS bounce.
// ---------------------------------------------------------------------------
__global__ __launch_bounds__(256)
void attn_kernel(const __bf16* __restrict__ qf, const __bf16* __restrict__ kf,
                 const __bf16* __restrict__ vT, __bf16* __restrict__ attn) {
  const int qt   = blockIdx.x;
  const int h    = blockIdx.y;
  const int tid  = threadIdx.x;
  const int wave = tid >> 5;
  const int lane = tid & 31;
  const int half = lane >> 4;
  const int nl   = lane & 15;

  __shared__ __bf16 sK[2][32][DQK];    // 24 KB
  __shared__ __bf16 sVT[2][DV][32];    // 16 KB  (V^T tile: d rows x 32 keys)
  __shared__ __bf16 sP[8][16][32];     //  8 KB

  const int qr0 = qt * 128 + wave * 16;

  v16bf fq[6];
  {
    const __bf16* qbase = qf + ((size_t)qr0 * NHEAD + h) * DQK;
#pragma unroll
    for (int c = 0; c < 6; ++c)
      fq[c] = load_frag(qbase + c * 32, NHEAD * DQK, lane);
  }

  v8f acc[8] = {};
  float mrow[8], lrow[8];
#pragma unroll
  for (int r = 0; r < 8; ++r) { mrow[r] = -1e30f; lrow[r] = 0.f; }

  const float scale = 0.07216878364870323f;   // 1/sqrt(192)
  const int ntiles = (qt + 1) * 4;            // 32-key tiles, causal bound

  const __bf16* vTh = vT + (size_t)h * DV * S_LEN;

  if (wave == 0) {
    tdm_load_2d_bf16(&sK[0][0][0],  kf + (size_t)h * DQK, DQK, 32, NHEAD * DQK);
    tdm_load_2d_bf16(&sVT[0][0][0], vTh,                  32, DV, S_LEN);
    __builtin_amdgcn_s_wait_tensorcnt(0);
  }
  __syncthreads();

  for (int t = 0; t < ntiles; ++t) {
    const int kt  = t * 32;
    const int cur = t & 1;
    const int nxt = cur ^ 1;
    if (wave == 0 && t + 1 < ntiles) {
      tdm_load_2d_bf16(&sK[nxt][0][0],
                       kf + ((size_t)(kt + 32) * NHEAD + h) * DQK, DQK, 32, NHEAD * DQK);
      tdm_load_2d_bf16(&sVT[nxt][0][0], vTh + kt + 32, 32, DV, S_LEN);
    }

    // ---- scores for 2 key sub-tiles (16 keys each)
    v8f s0 = {}, s1 = {};
#pragma unroll
    for (int c = 0; c < 6; ++c) {
      const v16bf fb0 = load_frag(&sK[cur][0][c * 32],  DQK, lane);
      const v16bf fb1 = load_frag(&sK[cur][16][c * 32], DQK, lane);
      s0 = __builtin_amdgcn_wmma_f32_16x16x32_bf16(false, fq[c], false, fb0,
                                                   (short)0, s0, false, false);
      s1 = __builtin_amdgcn_wmma_f32_16x16x32_bf16(false, fq[c], false, fb1,
                                                   (short)0, s1, false, false);
    }

    // ---- online softmax (row stats over the 16-lane N groups)
#pragma unroll
    for (int r = 0; r < 8; ++r) {
      const int row = qr0 + half * 8 + r;
      const int c0 = kt + nl, c1 = c0 + 16;
      const float e0 = (c0 <= row) ? s0[r] * scale : -1e30f;
      const float e1 = (c1 <= row) ? s1[r] * scale : -1e30f;
      float mx = fmaxf(e0, e1);
#pragma unroll
      for (int off = 1; off < 16; off <<= 1)
        mx = fmaxf(mx, __shfl_xor(mx, off, 32));
      const float mnew = fmaxf(mrow[r], mx);
      const float corr = __expf(mrow[r] - mnew);
      const float p0 = __expf(e0 - mnew);
      const float p1 = __expf(e1 - mnew);
      float rs = p0 + p1;
#pragma unroll
      for (int off = 1; off < 16; off <<= 1)
        rs += __shfl_xor(rs, off, 32);
      mrow[r] = mnew;
      lrow[r] = lrow[r] * corr + rs;
#pragma unroll
      for (int j = 0; j < 8; ++j) acc[j][r] *= corr;
      sP[wave][half * 8 + r][nl]      = (__bf16)p0;
      sP[wave][half * 8 + r][16 + nl] = (__bf16)p1;
    }

    // ---- PV: out[16x128] += P[16x32] * V[32x128]
    //      B-frag of V == A-style frag of V^T tile (row-major, contiguous)
    const v16bf fp = load_frag(&sP[wave][0][0], 32, lane);
#pragma unroll
    for (int j = 0; j < 8; ++j) {
      const v16bf fv = load_frag(&sVT[cur][j * 16][0], 32, lane);
      acc[j] = __builtin_amdgcn_wmma_f32_16x16x32_bf16(false, fp, false, fv,
                                                       (short)0, acc[j], false, false);
    }

    if (wave == 0) __builtin_amdgcn_s_wait_tensorcnt(0);
    __syncthreads();
  }

  // ---- normalize & write attn[S][H*128] bf16
#pragma unroll
  for (int r = 0; r < 8; ++r) {
    const int row = qr0 + half * 8 + r;
    const float invl = 1.0f / lrow[r];
#pragma unroll
    for (int j = 0; j < 8; ++j)
      attn[(size_t)row * (NHEAD * DV) + h * DV + j * 16 + nl] =
          (__bf16)(acc[j][r] * invl);
  }
}

// ---------------------------------------------------------------------------
// Host-side launch sequence
// ---------------------------------------------------------------------------
extern "C" void kernel_launch(void* const* d_in, const int* in_sizes, int n_in,
                              void* d_out, int out_size, void* d_ws, size_t ws_size,
                              hipStream_t stream) {
  (void)in_sizes; (void)n_in; (void)out_size; (void)ws_size;

  const int*   pos       = (const int*)d_in[0];
  const float* hs        = (const float*)d_in[1];
  const float* w_fused_a = (const float*)d_in[2];
  const float* q_ln      = (const float*)d_in[3];
  const float* kv_ln     = (const float*)d_in[4];
  const float* w_q_b     = (const float*)d_in[5];
  const float* w_kv_b    = (const float*)d_in[6];
  const float* w_o       = (const float*)d_in[7];
  float*       out       = (float*)d_out;

  char* ws = (char*)d_ws;
  size_t off = 0;
  auto alloc = [&](size_t bytes) -> void* {
    void* p = ws + off;
    off += (bytes + 255) & ~(size_t)255;
    return p;
  };
  // f32 intermediates
  float*  a_buf   = (float*)alloc((size_t)S_LEN * 2112 * 4);
  float*  q_buf   = (float*)alloc((size_t)S_LEN * NHEAD * DQK * 4);
  float*  kv_buf  = (float*)alloc((size_t)S_LEN * NHEAD * 256 * 4);
  // bf16 operand copies
  __bf16* hs_bf   = (__bf16*)alloc((size_t)S_LEN * HID * 2);
  __bf16* wfa_bf  = (__bf16*)alloc((size_t)2112 * HID * 2);
  __bf16* wqb_bf  = (__bf16*)alloc((size_t)6144 * 1536 * 2);
  __bf16* wkvb_bf = (__bf16*)alloc((size_t)8192 * 512 * 2);
  __bf16* wo_bf   = (__bf16*)alloc((size_t)HID * HID * 2);
  __bf16* cq_bf   = (__bf16*)alloc((size_t)S_LEN * 1536 * 2);
  __bf16* ckv_bf  = (__bf16*)alloc((size_t)S_LEN * 512 * 2);
  __bf16* qfb     = (__bf16*)alloc((size_t)S_LEN * NHEAD * DQK * 2);
  __bf16* kfb     = (__bf16*)alloc((size_t)S_LEN * NHEAD * DQK * 2);
  __bf16* vTb     = (__bf16*)alloc((size_t)NHEAD * DV * S_LEN * 2);
  __bf16* attn_bf = (__bf16*)alloc((size_t)S_LEN * NHEAD * DV * 2);

  const dim3 blk(256);
  auto cvt = [&](const float* src, __bf16* dst, long n) {
    cvt_bf16_kernel<<<(unsigned)(n / 1024), blk, 0, stream>>>(src, dst, n);
  };

  // 0) one-time bf16 conversions of activations + weights
  cvt(hs,        hs_bf,   (long)S_LEN * HID);
  cvt(w_fused_a, wfa_bf,  (long)2112 * HID);
  cvt(w_q_b,     wqb_bf,  (long)6144 * 1536);
  cvt(w_kv_b,    wkvb_bf, (long)8192 * 512);
  cvt(w_o,       wo_bf,   (long)HID * HID);

  // 1) fused A projection: [2048,4096] x [2112,4096]^T -> f32
  gemm_bf16<<<dim3(2112 / GT_N, S_LEN / GT_M), blk, 0, stream>>>(
      hs_bf, wfa_bf, a_buf, S_LEN, 2112, HID);

  // 2) RMS norms on latent slices (bf16 out)
  rmsnorm_kernel<<<S_LEN, blk, 0, stream>>>(a_buf, 2112, 0, 1536, q_ln, cq_bf);
  rmsnorm_kernel<<<S_LEN, blk, 0, stream>>>(a_buf, 2112, 1536, 512, kv_ln, ckv_bf);

  // 3) Q up-projection: [2048,1536] x [6144,1536]^T -> f32
  gemm_bf16<<<dim3(6144 / GT_N, S_LEN / GT_M), blk, 0, stream>>>(
      cq_bf, wqb_bf, q_buf, S_LEN, 6144, 1536);

  // 4) KV up-projection: [2048,512] x [8192,512]^T -> f32
  gemm_bf16<<<dim3(8192 / GT_N, S_LEN / GT_M), blk, 0, stream>>>(
      ckv_bf, wkvb_bf, kv_buf, S_LEN, 8192, 512);

  // 5) RoPE + pack bf16 attention operands (q/k row-major, V transposed)
  prep_qf_kernel<<<S_LEN, blk, 0, stream>>>(q_buf, pos, qfb);
  prep_kf_kernel<<<S_LEN, blk, 0, stream>>>(kv_buf, a_buf, pos, kfb);
  prep_vT_kernel<<<NHEAD * DV, blk, 0, stream>>>(kv_buf, vTb);

  // 6) causal flash attention (bf16 out)
  attn_kernel<<<dim3(S_LEN / 128, NHEAD), blk, 0, stream>>>(qfb, kfb, vTb, attn_bf);

  // 7) output projection: [2048,4096] x [4096,4096]^T -> d_out (f32)
  gemm_bf16<<<dim3(HID / GT_N, S_LEN / GT_M), blk, 0, stream>>>(
      attn_bf, wo_bf, out, S_LEN, HID, HID);
}
